// EncoderBlock_70411693850670
// MI455X (gfx1250) — matrix operand order repaired and checked by reading, not compiled, not run on hardware
//
#include <hip/hip_runtime.h>

#define D_MODEL 1024
#define NHEAD   16
#define DK      64
#define DFF     4096
#define SEQ     2048
#define BATCH   2
#define MTOT    (BATCH * SEQ) /* 4096 */

typedef __attribute__((ext_vector_type(16))) __bf16 v16bf;
typedef __attribute__((ext_vector_type(8)))  __bf16 v8bf;
typedef __attribute__((ext_vector_type(8)))  float  v8f;

__device__ __forceinline__ v8f wmma_bf16(v16bf a, v16bf b, v8f c) {
  // D = A(16x32) * B(32x16) + C, f32 accumulate
  return __builtin_amdgcn_wmma_f32_16x16x32_bf16(false, a, false, b, (short)0, c,
                                                 false, false);
}

// A fragment (16x32 bf16) from row-major [M, ld] source (global or LDS).
// ISA layout: lanes0-15 hold K 0-7 (v0-3) + 16-23 (v4-7); lanes16-31 hold K 8-15 + 24-31.
__device__ __forceinline__ v16bf load_a(const __bf16* base, int ld, int m0, int kb, int lane) {
  int r = lane & 15, hi = lane >> 4;
  const __bf16* p = base + (size_t)(m0 + r) * ld + kb + hi * 8;
  union { v16bf v; v8bf h[2]; } u;
  u.h[0] = *(const v8bf*)p;
  u.h[1] = *(const v8bf*)(p + 16);
  return u.v;
}

// B fragment (32x16 bf16) from [N, ld] (n-major) source, same K order as A.
// lanes0-15: n=lane, K=kb+0..7 + kb+16..23; lanes16-31: n=lane-16, K=kb+8..15 + kb+24..31.
__device__ __forceinline__ v16bf load_b(const __bf16* base, int ld, int n0, int kb, int lane) {
  int c = lane & 15, hi = lane >> 4;
  const __bf16* p = base + (size_t)(n0 + c) * ld + kb + hi * 8;
  union { v16bf v; v8bf h[2]; } u;
  u.h[0] = *(const v8bf*)p;
  u.h[1] = *(const v8bf*)(p + 16);
  return u.v;
}

// ---- CDNA5 async global->LDS copy (16B per lane), ASYNCcnt tracked ----
__device__ __forceinline__ void async_copy16(const void* gaddr, unsigned lds_off) {
  asm volatile("global_load_async_to_lds_b128 %0, %1, off"
               :: "v"(lds_off), "v"(gaddr)
               : "memory");
}

__device__ __forceinline__ void wait_async0() {
#if defined(__has_builtin)
#if __has_builtin(__builtin_amdgcn_s_wait_asynccnt)
  __builtin_amdgcn_s_wait_asynccnt(0);
#else
  asm volatile("s_wait_asynccnt 0" ::: "memory");
#endif
#else
  asm volatile("s_wait_asynccnt 0" ::: "memory");
#endif
}

// ---------------- fp32 -> bf16 cast ----------------
__global__ __launch_bounds__(256) void cast_bf16_kernel(const float* __restrict__ in,
                                                        __bf16* __restrict__ out, int n) {
  int i = (blockIdx.x * 256 + threadIdx.x) * 4;
  if (i + 3 < n) {
#pragma unroll
    for (int j = 0; j < 4; ++j) out[i + j] = (__bf16)in[i + j];
  }
}

// ---------------- LayerNorm (Bessel-corrected, eps added to std) ----------------
__global__ __launch_bounds__(256) void ln_kernel(const float* __restrict__ x,
                                                 const float* __restrict__ alpha,
                                                 const float* __restrict__ beta,
                                                 __bf16* __restrict__ out) {
  int row = blockIdx.x;
  const float* xr = x + (size_t)row * D_MODEL;
  __shared__ float red[2][8];
  float v[4], s = 0.f, ss = 0.f;
#pragma unroll
  for (int i = 0; i < 4; ++i) {
    v[i] = xr[threadIdx.x + i * 256];
    s += v[i];
    ss += v[i] * v[i];
  }
#pragma unroll
  for (int m = 16; m >= 1; m >>= 1) {
    s += __shfl_xor(s, m, 32);
    ss += __shfl_xor(ss, m, 32);
  }
  int wid = threadIdx.x >> 5;
  if ((threadIdx.x & 31) == 0) { red[0][wid] = s; red[1][wid] = ss; }
  __syncthreads();
  float ts = 0.f, tss = 0.f;
#pragma unroll
  for (int i = 0; i < 8; ++i) { ts += red[0][i]; tss += red[1][i]; }
  float mean = ts * (1.f / (float)D_MODEL);
  float var = (tss - (float)D_MODEL * mean * mean) * (1.f / (float)(D_MODEL - 1));
  var = fmaxf(var, 0.f);
  float inv = alpha[0] / (sqrtf(var) + 1e-5f);
  float b = beta[0];
  __bf16* orow = out + (size_t)row * D_MODEL;
#pragma unroll
  for (int i = 0; i < 4; ++i)
    orow[threadIdx.x + i * 256] = (__bf16)((v[i] - mean) * inv + b);
}

// ---------------- staged GEMM: C = A[M,K] * W[N,K]^T + bias ----------------
// Block tile 128(M) x 256(N), 8 waves in a 2x4 grid, wave tile 64x64.
// Per K=32 step: async-stage A(128x32)+B(256x32) into LDS (double buffered),
// fragments re-read from LDS, 16 WMMAs per wave per step.
// MODE 1: fp32 out = resid + acc + bias          (O-proj / FFN2 + residual)
// MODE 2: bf16 out = relu(acc + bias)            (FFN1)
// MODE 3: bf16 scatter to [B,H,S,DK]             (Q, K)
// MODE 4: bf16 scatter to [B,H,DK,S]             (V transposed)
__device__ __forceinline__ void stage_tile(const __bf16* __restrict__ A,
                                           const __bf16* __restrict__ W, int K,
                                           int bm0, int bn0, int kb,
                                           __bf16* As, __bf16* Bs, int tid) {
#pragma unroll
  for (int i = 0; i < 2; ++i) {          // A: 512 x 16B chunks
    int c = tid + i * 256;
    int row = c >> 2, c8 = (c & 3) * 8;
    async_copy16(A + (size_t)(bm0 + row) * K + kb + c8,
                 (unsigned)(size_t)(As + row * 32 + c8));
  }
#pragma unroll
  for (int i = 0; i < 4; ++i) {          // B: 1024 x 16B chunks
    int c = tid + i * 256;
    int n = c >> 2, c8 = (c & 3) * 8;
    async_copy16(W + (size_t)(bn0 + n) * K + kb + c8,
                 (unsigned)(size_t)(Bs + n * 32 + c8));
  }
}

template <int MODE>
__global__ __launch_bounds__(256) void gemm_kernel(const __bf16* __restrict__ A,
                                                   const __bf16* __restrict__ W,
                                                   const float* __restrict__ bias,
                                                   const float* __restrict__ resid,
                                                   void* __restrict__ outp,
                                                   int M, int N, int K) {
  __shared__ __align__(32) __bf16 As[2][128 * 32];
  __shared__ __align__(32) __bf16 Bs[2][256 * 32];
  int tid = threadIdx.x;
  int lane = tid & 31;
  int wave = tid >> 5;
  int wm = wave >> 2;            // 0..1
  int wn = wave & 3;             // 0..3
  int ntn = N >> 8;              // N / 256
  int bm0 = (blockIdx.x / ntn) << 7;
  int bn0 = (blockIdx.x % ntn) << 8;

  stage_tile(A, W, K, bm0, bn0, 0, As[0], Bs[0], tid);
  wait_async0();
  __syncthreads();

  v8f acc[4][4] = {};
  for (int kb = 0; kb < K; kb += 32) {
    int buf = (kb >> 5) & 1;
    bool more = (kb + 32 < K);
    if (more) stage_tile(A, W, K, bm0, bn0, kb + 32, As[buf ^ 1], Bs[buf ^ 1], tid);

    v16bf a[4], b[4];
#pragma unroll
    for (int i = 0; i < 4; ++i) a[i] = load_a(As[buf], 32, wm * 64 + 16 * i, 0, lane);
#pragma unroll
    for (int j = 0; j < 4; ++j) b[j] = load_b(Bs[buf], 32, wn * 64 + 16 * j, 0, lane);
#pragma unroll
    for (int i = 0; i < 4; ++i)
#pragma unroll
      for (int j = 0; j < 4; ++j) acc[i][j] = wmma_bf16(a[i], b[j], acc[i][j]);

    if (more) wait_async0();
    __syncthreads();
  }

  int m0 = bm0 + wm * 64, n0 = bn0 + wn * 64;
  int cl = lane & 15, hi = lane >> 4;
#pragma unroll
  for (int i = 0; i < 4; ++i) {
#pragma unroll
    for (int j = 0; j < 4; ++j) {
      int col = n0 + 16 * j + cl;
      float bv = bias[col];
#pragma unroll
      for (int r = 0; r < 8; ++r) {
        int row = m0 + 16 * i + r + hi * 8;
        float val = acc[i][j][r] + bv;
        if (MODE == 1) {
          size_t idx = (size_t)row * N + col;
          ((float*)outp)[idx] = resid[idx] + val;
        } else if (MODE == 2) {
          ((__bf16*)outp)[(size_t)row * N + col] = (__bf16)fmaxf(val, 0.f);
        } else if (MODE == 3) {
          int b_ = row >> 11, s_ = row & (SEQ - 1);
          int h_ = col >> 6, d_ = col & (DK - 1);
          ((__bf16*)outp)[(((size_t)(b_ * NHEAD + h_)) * SEQ + s_) * DK + d_] = (__bf16)val;
        } else if (MODE == 4) {
          int b_ = row >> 11, s_ = row & (SEQ - 1);
          int h_ = col >> 6, d_ = col & (DK - 1);
          ((__bf16*)outp)[(((size_t)(b_ * NHEAD + h_)) * DK + d_) * SEQ + s_] = (__bf16)val;
        }
      }
    }
  }
}

// ---------------- fused attention (flash-style, online softmax) ----------------
// One wave per 16-row Q tile of one (b,h). K blocks of 64 keys.
__global__ __launch_bounds__(128) void attn_kernel(const __bf16* __restrict__ q,
                                                   const __bf16* __restrict__ k,
                                                   const __bf16* __restrict__ vT,
                                                   const int* __restrict__ mask,
                                                   __bf16* __restrict__ aout) {
  __shared__ __align__(32) __bf16 plds[4][16 * 80];
  int wid = threadIdx.x >> 5;
  int lane = threadIdx.x & 31;
  int cl = lane & 15, hi = lane >> 4;
  int gw = blockIdx.x * 4 + wid;
  int qt = gw & (SEQ / 16 - 1);  // 128 q-tiles per head
  int bh = gw >> 7;
  int b_ = bh >> 4;
  const __bf16* qh = q + (size_t)bh * SEQ * DK;
  const __bf16* kh = k + (size_t)bh * SEQ * DK;
  const __bf16* vh = vT + (size_t)bh * DK * SEQ;
  const int* mrow = mask + (size_t)b_ * SEQ;
  __bf16* pl = plds[wid];

  v16bf qa0 = load_a(qh, DK, qt * 16, 0, lane);
  v16bf qa1 = load_a(qh, DK, qt * 16, 32, lane);

  v8f o[4] = {};
  float rm[8], rs[8];
#pragma unroll
  for (int r = 0; r < 8; ++r) { rm[r] = -1e30f; rs[r] = 0.f; }

  for (int j = 0; j < SEQ; j += 64) {
    // S tile = Q * K^T (K-dim 64 = 2 WMMA accumulations per 16x16 tile)
    v8f sc[4];
#pragma unroll
    for (int t = 0; t < 4; ++t) {
      v8f z = {};
      z = wmma_bf16(qa0, load_b(kh, DK, j + t * 16, 0, lane), z);
      z = wmma_bf16(qa1, load_b(kh, DK, j + t * 16, 32, lane), z);
      sc[t] = z;
    }
    // scale + mask, block row-max
    float bm[8];
#pragma unroll
    for (int r = 0; r < 8; ++r) bm[r] = -1e30f;
#pragma unroll
    for (int t = 0; t < 4; ++t) {
      bool live = (mrow[j + t * 16 + cl] != 0);
#pragma unroll
      for (int r = 0; r < 8; ++r) {
        float v = live ? sc[t][r] * 0.125f : -1e9f;
        sc[t][r] = v;
        bm[r] = fmaxf(bm[r], v);
      }
    }
#pragma unroll
    for (int m = 1; m < 16; m <<= 1)
#pragma unroll
      for (int r = 0; r < 8; ++r) bm[r] = fmaxf(bm[r], __shfl_xor(bm[r], m, 32));
    // online rescale
    float f[8];
#pragma unroll
    for (int r = 0; r < 8; ++r) {
      float nm = fmaxf(rm[r], bm[r]);
      f[r] = __expf(rm[r] - nm);
      rm[r] = nm;
    }
#pragma unroll
    for (int t = 0; t < 4; ++t)
#pragma unroll
      for (int r = 0; r < 8; ++r) o[t][r] *= f[r];
    // P = exp(S - rowmax), row sums
    float ps[8];
#pragma unroll
    for (int r = 0; r < 8; ++r) ps[r] = 0.f;
#pragma unroll
    for (int t = 0; t < 4; ++t)
#pragma unroll
      for (int r = 0; r < 8; ++r) {
        float p = __expf(sc[t][r] - rm[r]);
        sc[t][r] = p;
        ps[r] += p;
      }
#pragma unroll
    for (int m = 1; m < 16; m <<= 1)
#pragma unroll
      for (int r = 0; r < 8; ++r) ps[r] += __shfl_xor(ps[r], m, 32);
#pragma unroll
    for (int r = 0; r < 8; ++r) rs[r] = rs[r] * f[r] + ps[r];
    // C-layout -> A-layout via per-wave LDS tile (16x80 bf16, 16B-aligned rows)
#pragma unroll
    for (int t = 0; t < 4; ++t)
#pragma unroll
      for (int r = 0; r < 8; ++r)
        pl[(r + hi * 8) * 80 + t * 16 + cl] = (__bf16)sc[t][r];
    v16bf pa0 = load_a(pl, 80, 0, 0, lane);
    v16bf pa1 = load_a(pl, 80, 0, 32, lane);
    // O += P * V  (V stored transposed [DK, S] -> B frags contiguous)
#pragma unroll
    for (int t2 = 0; t2 < 4; ++t2) {
      o[t2] = wmma_bf16(pa0, load_b(vh, SEQ, t2 * 16, j, lane), o[t2]);
      o[t2] = wmma_bf16(pa1, load_b(vh, SEQ, t2 * 16, j + 32, lane), o[t2]);
    }
  }
  // normalize + write [B,S,D_MODEL] bf16
#pragma unroll
  for (int t2 = 0; t2 < 4; ++t2)
#pragma unroll
    for (int r = 0; r < 8; ++r) {
      int s_ = qt * 16 + r + hi * 8;
      int col = (bh & 15) * DK + t2 * 16 + cl;
      aout[((size_t)b_ * SEQ + s_) * D_MODEL + col] = (__bf16)(o[t2][r] / rs[r]);
    }
}

// ---------------- workspace layout (bytes) ----------------
#define OFF_XN1 ((size_t)0)                       // 4096*1024 bf16
#define OFF_Q   (OFF_XN1 + 8388608)
#define OFF_K   (OFF_Q + 8388608)
#define OFF_VT  (OFF_K + 8388608)
#define OFF_ATT (OFF_VT + 8388608)
#define OFF_XN2 (OFF_ATT + 8388608)
#define OFF_X1  (OFF_XN2 + 8388608)               // fp32
#define OFF_H1  (OFF_X1 + 16777216)               // 4096*4096 bf16
#define OFF_WQ  (OFF_H1 + 33554432)
#define OFF_WK  (OFF_WQ + 2097152)
#define OFF_WV  (OFF_WK + 2097152)
#define OFF_WO  (OFF_WV + 2097152)
#define OFF_W1  (OFF_WO + 2097152)
#define OFF_W2  (OFF_W1 + 8388608)

extern "C" void kernel_launch(void* const* d_in, const int* in_sizes, int n_in,
                              void* d_out, int out_size, void* d_ws, size_t ws_size,
                              hipStream_t stream) {
  (void)in_sizes; (void)n_in; (void)out_size; (void)ws_size;
  const float* x   = (const float*)d_in[0];
  const int*   mask= (const int*)d_in[1];
  const float* wq  = (const float*)d_in[2];
  const float* bq  = (const float*)d_in[3];
  const float* wk  = (const float*)d_in[4];
  const float* bk  = (const float*)d_in[5];
  const float* wv  = (const float*)d_in[6];
  const float* bv  = (const float*)d_in[7];
  const float* wo  = (const float*)d_in[8];
  const float* bo  = (const float*)d_in[9];
  const float* w1  = (const float*)d_in[10];
  const float* b1  = (const float*)d_in[11];
  const float* w2  = (const float*)d_in[12];
  const float* b2  = (const float*)d_in[13];
  const float* a1  = (const float*)d_in[14];
  const float* n1  = (const float*)d_in[15];
  const float* a2  = (const float*)d_in[16];
  const float* n2  = (const float*)d_in[17];

  char* ws = (char*)d_ws;
  __bf16* xn1 = (__bf16*)(ws + OFF_XN1);
  __bf16* qb  = (__bf16*)(ws + OFF_Q);
  __bf16* kb  = (__bf16*)(ws + OFF_K);
  __bf16* vtb = (__bf16*)(ws + OFF_VT);
  __bf16* att = (__bf16*)(ws + OFF_ATT);
  __bf16* xn2 = (__bf16*)(ws + OFF_XN2);
  float*  x1  = (float*)(ws + OFF_X1);
  __bf16* h1  = (__bf16*)(ws + OFF_H1);
  __bf16* wqb = (__bf16*)(ws + OFF_WQ);
  __bf16* wkb = (__bf16*)(ws + OFF_WK);
  __bf16* wvb = (__bf16*)(ws + OFF_WV);
  __bf16* wob = (__bf16*)(ws + OFF_WO);
  __bf16* w1b = (__bf16*)(ws + OFF_W1);
  __bf16* w2b = (__bf16*)(ws + OFF_W2);

  const int nW = D_MODEL * D_MODEL;       // 1,048,576
  const int nF = DFF * D_MODEL;           // 4,194,304
  cast_bf16_kernel<<<nW / 1024, 256, 0, stream>>>(wq, wqb, nW);
  cast_bf16_kernel<<<nW / 1024, 256, 0, stream>>>(wk, wkb, nW);
  cast_bf16_kernel<<<nW / 1024, 256, 0, stream>>>(wv, wvb, nW);
  cast_bf16_kernel<<<nW / 1024, 256, 0, stream>>>(wo, wob, nW);
  cast_bf16_kernel<<<nF / 1024, 256, 0, stream>>>(w1, w1b, nF);
  cast_bf16_kernel<<<nF / 1024, 256, 0, stream>>>(w2, w2b, nF);

  // LN1
  ln_kernel<<<MTOT, 256, 0, stream>>>(x, a1, n1, xn1);

  // QKV projections: 4096x1024x1024; (4096/128)x(1024/256) = 128 blocks
  gemm_kernel<3><<<128, 256, 0, stream>>>(xn1, wqb, bq, nullptr, qb, MTOT, D_MODEL, D_MODEL);
  gemm_kernel<3><<<128, 256, 0, stream>>>(xn1, wkb, bk, nullptr, kb, MTOT, D_MODEL, D_MODEL);
  gemm_kernel<4><<<128, 256, 0, stream>>>(xn1, wvb, bv, nullptr, vtb, MTOT, D_MODEL, D_MODEL);

  // fused attention: 2*16*128 = 4096 waves / 4 per block
  attn_kernel<<<1024, 128, 0, stream>>>(qb, kb, vtb, mask, att);

  // O projection + residual -> x1 (fp32)
  gemm_kernel<1><<<128, 256, 0, stream>>>(att, wob, bo, x, x1, MTOT, D_MODEL, D_MODEL);

  // LN2
  ln_kernel<<<MTOT, 256, 0, stream>>>(x1, a2, n2, xn2);

  // FFN1: 4096x4096x1024, relu -> bf16; 32x16 = 512 blocks
  gemm_kernel<2><<<512, 256, 0, stream>>>(xn2, w1b, b1, nullptr, h1, MTOT, DFF, D_MODEL);

  // FFN2 + residual -> d_out (fp32): 4096x1024x4096
  gemm_kernel<1><<<128, 256, 0, stream>>>(h1, w2b, b2, x1, d_out, MTOT, D_MODEL, DFF);
}
